// SmallthinkerMoeBlock_62560493633733
// MI455X (gfx1250) — compile-verified, matrix operand order checked by reading, not run on hardware
//
#include <hip/hip_runtime.h>
#include <math.h>

// ---------------- problem constants ----------------
#define T_TOK 4096
#define HDIM  2048
#define EDIM  32
#define FDIM  768
#define KTOP  4
#define NSG   8
#define SZG   96
#define CAPC  1024
#define KT    32       // K step per iteration (one WMMA K-step, double-buffered)
#define BPAD  40       // padded LDS row stride (elements) -> 80B, 16B-aligned, conflict-free

typedef __attribute__((ext_vector_type(16))) __bf16 v16bf;
typedef __attribute__((ext_vector_type(8)))  __bf16 v8bf;
typedef __attribute__((ext_vector_type(4)))  __bf16 v4bf;
typedef __attribute__((ext_vector_type(8)))  float  v8f;

// ---------------- workspace layout (bytes) ----------------
#define OFF_SEL 0
#define OFF_RW  (OFF_SEL + (size_t)T_TOK * KTOP * 4)
#define OFF_IDX (OFF_RW  + (size_t)T_TOK * KTOP * 4)
#define OFF_W   (OFF_IDX + (size_t)EDIM * CAPC * 4)
#define OFF_SG  (OFF_W   + (size_t)EDIM * CAPC * 4)
#define OFF_ACT (OFF_SG  + (size_t)EDIM * CAPC)                    // 256B-aligned
#define OFF_XB  (OFF_ACT + (size_t)EDIM * CAPC * FDIM * 2)         // x in bf16

// Load one 16x32 bf16 WMMA fragment from a row pointer.
// Lane l<16 -> row, halves at K = hi*8 and 16+hi*8 (A-layout; B mirrored with lane=N).
static __device__ inline v16bf load_frag(const __bf16* rowp, int hi) {
    v8bf a0 = *(const v8bf*)(rowp + hi * 8);
    v8bf a1 = *(const v8bf*)(rowp + 16 + hi * 8);
    return __builtin_shufflevector(a0, a1, 0,1,2,3,4,5,6,7,8,9,10,11,12,13,14,15);
}

// =====================================================================
// 0a) zero the token-accumulation region of d_out
// =====================================================================
__global__ void zero_kernel(float* __restrict__ p, long n) {
    long i = (long)blockIdx.x * blockDim.x + threadIdx.x;
    long stride = (long)gridDim.x * blockDim.x;
    for (; i < n; i += stride) p[i] = 0.f;
}

// =====================================================================
// 0b) pre-convert hidden_states to bf16 (read once; halves A-side traffic)
// =====================================================================
__global__ void cvt_kernel(const float* __restrict__ x, __bf16* __restrict__ xb, long n) {
    long i = ((long)blockIdx.x * blockDim.x + threadIdx.x) * 4;
    long stride = (long)gridDim.x * blockDim.x * 4;
    for (; i < n; i += stride) {
        float4 f = *(const float4*)(x + i);
        v4bf c = { (__bf16)f.x, (__bf16)f.y, (__bf16)f.z, (__bf16)f.w };
        *(v4bf*)(xb + i) = c;
    }
}

// =====================================================================
// 1) router GEMV + stable top-4 + softmax.  One wave per token.
// =====================================================================
__global__ __launch_bounds__(32) void router_kernel(
    const float* __restrict__ r, const float* __restrict__ wr,
    int* __restrict__ sel, float* __restrict__ rw, float* __restrict__ toplog)
{
    const int t    = blockIdx.x;
    const int lane = threadIdx.x;            // == expert index
    const float* rr = r + (long)t * HDIM;

    float acc = 0.f;
    for (int h = 0; h < HDIM; h += 4) {
        float4 rv = *(const float4*)(rr + h);
        acc += rv.x * wr[(h + 0) * EDIM + lane];
        acc += rv.y * wr[(h + 1) * EDIM + lane];
        acc += rv.z * wr[(h + 2) * EDIM + lane];
        acc += rv.w * wr[(h + 3) * EDIM + lane];
    }

    float v = acc;
    float tl[KTOP]; int ts[KTOP];
#pragma unroll
    for (int k = 0; k < KTOP; k++) {
        float m = v;
        for (int off = 16; off >= 1; off >>= 1) m = fmaxf(m, __shfl_xor(m, off, 32));
        unsigned bal = (unsigned)__ballot(v == m);
        int src = __builtin_ctz(bal);        // lowest expert wins ties (stable top_k)
        tl[k] = m; ts[k] = src;
        if (lane == src) v = -INFINITY;
    }

    if (lane == 0) {
        float mx = tl[0], s = 0.f, ex[KTOP];
#pragma unroll
        for (int k = 0; k < KTOP; k++) { ex[k] = __expf(tl[k] - mx); s += ex[k]; }
        float inv = 1.f / s;
#pragma unroll
        for (int k = 0; k < KTOP; k++) {
            toplog[t * KTOP + k] = tl[k];
            sel[t * KTOP + k]    = ts[k];
            rw[t * KTOP + k]     = ex[k] * inv;
        }
    }
}

// =====================================================================
// 2) dispatch: per expert, first CAP routed tokens in token order.
// =====================================================================
__global__ __launch_bounds__(32) void dispatch_kernel(
    const int* __restrict__ sel, const float* __restrict__ rw,
    int* __restrict__ idxo, float* __restrict__ wo)
{
    const int e    = blockIdx.x;
    const int lane = threadIdx.x;
    int count = 0;
    for (int base = 0; base < T_TOK; base += 32) {
        const int t = base + lane;
        bool hit = false; float wv = 0.f;
#pragma unroll
        for (int k = 0; k < KTOP; k++)
            if (sel[t * KTOP + k] == e) { hit = true; wv = rw[t * KTOP + k]; }
        unsigned m32 = (unsigned)__ballot(hit);
        int pos = count + __popc(m32 & ((1u << lane) - 1u));
        if (hit && pos < CAPC) { idxo[e * CAPC + pos] = t; wo[e * CAPC + pos] = wv; }
        count += __popc(m32);
    }
    for (int p = count + lane; p < CAPC; p += 32) {  // padding: weight 0 => zero contribution
        idxo[e * CAPC + p] = 0; wo[e * CAPC + p] = 0.f;
    }
}

// =====================================================================
// 3) secondary gate bits: sigmoid(r·w_sgate) > 0.5  <=>  z > 0
// =====================================================================
__global__ __launch_bounds__(256) void sgate_kernel(
    const float* __restrict__ r, const int* __restrict__ idx,
    const float* __restrict__ wsg, unsigned char* __restrict__ sg)
{
    const int slot = blockIdx.x * 8 + (threadIdx.x >> 5);
    const int lane = threadIdx.x & 31;
    const int e = slot / CAPC;
    const int t = idx[slot];
    const float* rr = r + (long)t * HDIM;
    const float* wb = wsg + (long)e * HDIM * NSG;

    float p[NSG] = {};
    for (int h = lane; h < HDIM; h += 32) {
        float rv = rr[h];
        const float* wp = wb + (long)h * NSG;
#pragma unroll
        for (int s = 0; s < NSG; s++) p[s] += rv * wp[s];
    }
#pragma unroll
    for (int s = 0; s < NSG; s++)
        for (int off = 16; off >= 1; off >>= 1) p[s] += __shfl_xor(p[s], off, 32);
    if (lane == 0) {
        unsigned b = 0;
#pragma unroll
        for (int s = 0; s < NSG; s++) if (p[s] > 0.f) b |= (1u << s);
        sg[slot] = (unsigned char)b;
    }
}

// ---------------------------------------------------------------------
// Stage a 32(K) x 128(N) fp32 weight slab into LDS as bf16 [N][K].
// 512 threads: bn0 = (tid&63)*2 (n fastest -> coalesced 256B/row per wave),
// bkq = (tid>>6)*4; pack 4 K per column -> ds_store_b64.
// ---------------------------------------------------------------------
static __device__ inline void stage_b32(__bf16 (*sB)[BPAD], const float* src,
                                        long row_stride, int bn0, int bkq) {
    float rv[4][2];
#pragma unroll
    for (int rr = 0; rr < 4; rr++) {
        float2 a = *(const float2*)(src + (long)rr * row_stride);
        rv[rr][0] = a.x; rv[rr][1] = a.y;
    }
#pragma unroll
    for (int j = 0; j < 2; j++) {
        v4bf c = { (__bf16)rv[0][j], (__bf16)rv[1][j], (__bf16)rv[2][j], (__bf16)rv[3][j] };
        *(v4bf*)&sB[bn0 + j][bkq] = c;
    }
}

// =====================================================================
// 4) up/gate GEMM (bf16 WMMA) + relu + group-mask -> act (bf16)
//    Block tile 128(M) x 128(N); 16 waves in 4x4 grid, 32x32 per wave;
//    K step 32, double-buffered LDS (one barrier per step).
// =====================================================================
__global__ __launch_bounds__(512) void upgate_kernel(
    const __bf16* __restrict__ xb, const int* __restrict__ idx,
    const unsigned char* __restrict__ sg,
    const float* __restrict__ wup, const float* __restrict__ wgate,
    __bf16* __restrict__ act)
{
    __shared__ __attribute__((aligned(16))) __bf16 sA[2][128][BPAD];
    __shared__ __attribute__((aligned(16))) __bf16 sBu[2][128][BPAD];
    __shared__ __attribute__((aligned(16))) __bf16 sBg[2][128][BPAD];

    const int NT = FDIM / 128;               // 6
    const int e  = blockIdx.x / ((CAPC / 128) * NT);
    const int rm = blockIdx.x % ((CAPC / 128) * NT);
    const int mt = rm / NT;
    const int nt = rm % NT;

    const int tid  = threadIdx.x;
    const int lane = tid & 31;
    const int wave = tid >> 5;               // 0..15
    const int l    = lane & 15;
    const int hi   = lane >> 4;
    const int wm   = wave & 3;               // M slab (32 rows)
    const int wn   = wave >> 2;              // N slab (32 cols)

    // A staging: 4 threads per row, 8 bf16 cols each (pure b128 copy)
    const int arow  = tid >> 2;              // 0..127
    const int apart = tid & 3;
    const long t_st = idx[e * CAPC + mt * 128 + arow];
    const __bf16* xrow = xb + t_st * (long)HDIM + apart * 8;

    // B staging map
    const int bn0 = (tid & 63) * 2;
    const int bkq = (tid >> 6) * 4;
    const long wb = (long)e * HDIM * FDIM + (long)(nt * 128 + bn0);

    v8f accu[2][2] = {}; v8f accg[2][2] = {};

    // prologue: stage slab 0 into buffer 0
    {
        *(v8bf*)&sA[0][arow][apart * 8] = *(const v8bf*)(xrow);
        stage_b32(sBu[0], wup   + wb + (long)bkq * FDIM, FDIM, bn0, bkq);
        stage_b32(sBg[0], wgate + wb + (long)bkq * FDIM, FDIM, bn0, bkq);
    }

    int p = 0;
    for (int k0 = 0; k0 < HDIM; k0 += KT) {
        __syncthreads();                      // staging of buffer p complete
        if (k0 + KT < HDIM) {                 // stage next slab into p^1 (overlaps WMMAs)
            const int q = p ^ 1;
            const float* pu = wup   + wb + (long)(k0 + KT + bkq) * FDIM;
            const float* pg = wgate + wb + (long)(k0 + KT + bkq) * FDIM;
            __builtin_prefetch(pu + (long)KT * FDIM, 0, 0);   // global_prefetch_b8
            __builtin_prefetch(pg + (long)KT * FDIM, 0, 0);
            *(v8bf*)&sA[q][arow][apart * 8] = *(const v8bf*)(xrow + k0 + KT);
            stage_b32(sBu[q], pu, FDIM, bn0, bkq);
            stage_b32(sBg[q], pg, FDIM, bn0, bkq);
        }

        v16bf a0 = load_frag(&sA[p][wm * 32 + l][0],       hi);
        v16bf a1 = load_frag(&sA[p][wm * 32 + 16 + l][0],  hi);
        v16bf u0 = load_frag(&sBu[p][wn * 32 + l][0],      hi);
        v16bf u1 = load_frag(&sBu[p][wn * 32 + 16 + l][0], hi);
        v16bf g0 = load_frag(&sBg[p][wn * 32 + l][0],      hi);
        v16bf g1 = load_frag(&sBg[p][wn * 32 + 16 + l][0], hi);

        accu[0][0] = __builtin_amdgcn_wmma_f32_16x16x32_bf16(false, a0, false, u0, (short)0, accu[0][0], false, false);
        accu[0][1] = __builtin_amdgcn_wmma_f32_16x16x32_bf16(false, a0, false, u1, (short)0, accu[0][1], false, false);
        accu[1][0] = __builtin_amdgcn_wmma_f32_16x16x32_bf16(false, a1, false, u0, (short)0, accu[1][0], false, false);
        accu[1][1] = __builtin_amdgcn_wmma_f32_16x16x32_bf16(false, a1, false, u1, (short)0, accu[1][1], false, false);
        accg[0][0] = __builtin_amdgcn_wmma_f32_16x16x32_bf16(false, a0, false, g0, (short)0, accg[0][0], false, false);
        accg[0][1] = __builtin_amdgcn_wmma_f32_16x16x32_bf16(false, a0, false, g1, (short)0, accg[0][1], false, false);
        accg[1][0] = __builtin_amdgcn_wmma_f32_16x16x32_bf16(false, a1, false, g0, (short)0, accg[1][0], false, false);
        accg[1][1] = __builtin_amdgcn_wmma_f32_16x16x32_bf16(false, a1, false, g1, (short)0, accg[1][1], false, false);

        p ^= 1;
    }

    // epilogue: act = up * relu(gate) * group_bit -> bf16
#pragma unroll
    for (int mi = 0; mi < 2; mi++) {
#pragma unroll
        for (int ni = 0; ni < 2; ni++) {
            const int f   = nt * 128 + wn * 32 + ni * 16 + l;
            const int grp = f / SZG;
#pragma unroll
            for (int vi = 0; vi < 8; vi++) {
                const int c = mt * 128 + wm * 32 + mi * 16 + hi * 8 + vi;  // M = vi + 8*hi
                const unsigned char bits = sg[e * CAPC + c];
                float g = accg[mi][ni][vi]; g = g > 0.f ? g : 0.f;
                float val = accu[mi][ni][vi] * g;
                val = ((bits >> grp) & 1) ? val : 0.f;
                act[((long)(e * CAPC + c)) * FDIM + f] = (__bf16)val;
            }
        }
    }
}

// =====================================================================
// 5) down GEMM (bf16 WMMA) + routing-weight scale + atomic scatter
//    Block tile 128x128; A direct from bf16 act; B double-buffered.
// =====================================================================
__global__ __launch_bounds__(512) void down_kernel(
    const __bf16* __restrict__ act, const int* __restrict__ idx,
    const float* __restrict__ wslot, const float* __restrict__ wdown,
    float* __restrict__ out)
{
    __shared__ __attribute__((aligned(16))) __bf16 sB[2][128][BPAD];

    const int NT = HDIM / 128;               // 16
    const int e  = blockIdx.x / ((CAPC / 128) * NT);
    const int rm = blockIdx.x % ((CAPC / 128) * NT);
    const int mt = rm / NT;
    const int nt = rm % NT;

    const int tid  = threadIdx.x;
    const int lane = tid & 31;
    const int wave = tid >> 5;
    const int l    = lane & 15;
    const int hi   = lane >> 4;
    const int wm   = wave & 3;
    const int wn   = wave >> 2;

    const int bn0 = (tid & 63) * 2;
    const int bkq = (tid >> 6) * 4;
    const long wb = (long)e * FDIM * HDIM + (long)(nt * 128 + bn0);

    // A loads straight from bf16 act (K-contiguous in memory)
    const __bf16* arp0 = act + ((long)(e * CAPC + mt * 128 + wm * 32 + l)) * FDIM;
    const __bf16* arp1 = act + ((long)(e * CAPC + mt * 128 + wm * 32 + 16 + l)) * FDIM;

    v8f acc[2][2] = {};

    stage_b32(sB[0], wdown + wb + (long)bkq * HDIM, HDIM, bn0, bkq);

    int p = 0;
    for (int k0 = 0; k0 < FDIM; k0 += KT) {
        __syncthreads();
        if (k0 + KT < FDIM) {
            const float* pd = wdown + wb + (long)(k0 + KT + bkq) * HDIM;
            __builtin_prefetch(pd + (long)KT * HDIM, 0, 0);
            stage_b32(sB[p ^ 1], pd, HDIM, bn0, bkq);
        }

        v16bf a0 = load_frag(arp0 + k0, hi);
        v16bf a1 = load_frag(arp1 + k0, hi);
        v16bf b0 = load_frag(&sB[p][wn * 32 + l][0],      hi);
        v16bf b1 = load_frag(&sB[p][wn * 32 + 16 + l][0], hi);

        acc[0][0] = __builtin_amdgcn_wmma_f32_16x16x32_bf16(false, a0, false, b0, (short)0, acc[0][0], false, false);
        acc[0][1] = __builtin_amdgcn_wmma_f32_16x16x32_bf16(false, a0, false, b1, (short)0, acc[0][1], false, false);
        acc[1][0] = __builtin_amdgcn_wmma_f32_16x16x32_bf16(false, a1, false, b0, (short)0, acc[1][0], false, false);
        acc[1][1] = __builtin_amdgcn_wmma_f32_16x16x32_bf16(false, a1, false, b1, (short)0, acc[1][1], false, false);

        p ^= 1;
    }

    // epilogue: scale by routing weight, scatter-accumulate to tokens
#pragma unroll
    for (int mi = 0; mi < 2; mi++) {
#pragma unroll
        for (int ni = 0; ni < 2; ni++) {
            const int n = nt * 128 + wn * 32 + ni * 16 + l;
#pragma unroll
            for (int vi = 0; vi < 8; vi++) {
                const int c = mt * 128 + wm * 32 + mi * 16 + hi * 8 + vi;
                const float wv = wslot[e * CAPC + c];
                if (wv != 0.f) {
                    const int t = idx[e * CAPC + c];
                    unsafeAtomicAdd(out + (long)t * HDIM + n, acc[mi][ni][vi] * wv);
                }
            }
        }
    }
}

// =====================================================================
// launch
// =====================================================================
extern "C" void kernel_launch(void* const* d_in, const int* in_sizes, int n_in,
                              void* d_out, int out_size, void* d_ws, size_t ws_size,
                              hipStream_t stream) {
    const float* r     = (const float*)d_in[0];   // router_input
    const float* x     = (const float*)d_in[1];   // hidden_states
    const float* wr    = (const float*)d_in[2];   // w_router [H,E]
    const float* wup   = (const float*)d_in[3];   // w_up   [E,H,F]
    const float* wgate = (const float*)d_in[4];   // w_gate [E,H,F]
    const float* wdown = (const float*)d_in[5];   // w_down [E,F,H]
    const float* wsg   = (const float*)d_in[6];   // w_sgate[E,H,NS]

    float* out    = (float*)d_out;                        // final [T,H]
    float* toplog = out + (long)T_TOK * HDIM;             // top_logits [T,K]

    char* ws = (char*)d_ws;
    int*           sel   = (int*)(ws + OFF_SEL);
    float*         rw    = (float*)(ws + OFF_RW);
    int*           idx   = (int*)(ws + OFF_IDX);
    float*         wslot = (float*)(ws + OFF_W);
    unsigned char* sg    = (unsigned char*)(ws + OFF_SG);
    __bf16*        act   = (__bf16*)(ws + OFF_ACT);
    __bf16*        xb    = (__bf16*)(ws + OFF_XB);

    zero_kernel<<<1024, 256, 0, stream>>>(out, (long)T_TOK * HDIM);
    cvt_kernel<<<1024, 256, 0, stream>>>(x, xb, (long)T_TOK * HDIM);
    router_kernel<<<T_TOK, 32, 0, stream>>>(r, wr, sel, rw, toplog);
    dispatch_kernel<<<EDIM, 32, 0, stream>>>(sel, rw, idx, wslot);
    sgate_kernel<<<(EDIM * CAPC) / 8, 256, 0, stream>>>(r, idx, wsg, sg);
    upgate_kernel<<<EDIM * (CAPC / 128) * (FDIM / 128), 512, 0, stream>>>(
        xb, idx, sg, wup, wgate, act);
    down_kernel<<<EDIM * (CAPC / 128) * (HDIM / 128), 512, 0, stream>>>(
        act, idx, wslot, wdown, out);
}
